// LSTM_3212635538049
// MI455X (gfx1250) — matrix operand order repaired and checked by reading, not compiled, not run on hardware
//
#include <hip/hip_runtime.h>

typedef __attribute__((ext_vector_type(2)))  float  v2f;
typedef __attribute__((ext_vector_type(8)))  float  v8f;
typedef __attribute__((ext_vector_type(8)))  __bf16 v8bf;
typedef __attribute__((ext_vector_type(16))) __bf16 v16bf;

#define Bsz 128
#define Tsz 1024
#define Psz 128
#define G4P 512   // 4*P
#define DIN 64
#define DOUT 64

__device__ __forceinline__ float sigmoidf_(float x) {
    return 1.0f / (1.0f + __expf(-x));
}
__device__ __forceinline__ float tanhf_(float x) {
    x = fminf(fmaxf(x, -20.0f), 20.0f);
    float e = __expf(2.0f * x);
    return (e - 1.0f) / (e + 1.0f);
}

// ---------------------------------------------------------------------------
// Prep: w_hh -> bf16 (row-major [512,128]), bias = b_ih + b_hh per layer
// ---------------------------------------------------------------------------
__global__ void prep_kernel(const float* __restrict__ wh0, const float* __restrict__ wh1,
                            const float* __restrict__ b_ih, const float* __restrict__ b_hh,
                            __bf16* __restrict__ wb0, __bf16* __restrict__ wb1,
                            float* __restrict__ bias0, float* __restrict__ bias1) {
    int i = blockIdx.x * 256 + threadIdx.x;
    if (i < G4P * Psz) {
        wb0[i] = (__bf16)wh0[i];
        wb1[i] = (__bf16)wh1[i];
    }
    if (i < G4P) {
        bias0[i] = b_ih[i] + b_hh[i];
        bias1[i] = b_ih[G4P + i] + b_hh[G4P + i];
    }
}

// ---------------------------------------------------------------------------
// Feed-forward GEMM: gx[t*B+b][n] = sum_k A_row(k) * W[n][k] + bias[n]
// f32 WMMA 16x16x4 (exact). Block = 8 waves; each wave owns 4 N-tiles (64
// gate columns) sharing one A fragment -> 4 independent WMMA chains.
// grid.x = (T*B)/16 (M tiles). N = 512 covered by the 8 waves.
// a_is_btd: A is x laid out [B,T,DIN]; else A is h laid out [T,B,P].
// gx stores are non-temporal (268 MB streamed once; keep L2 for h).
// ---------------------------------------------------------------------------
__global__ void __launch_bounds__(256) gemm_gx_kernel(const float* __restrict__ A,
                                                      const float* __restrict__ W,
                                                      const float* __restrict__ bias,
                                                      float* __restrict__ gx,
                                                      int K, int a_is_btd) {
    const int lane = threadIdx.x & 31;
    const int wave = threadIdx.x >> 5;
    const int l16  = lane & 15;
    const int hi   = lane >> 4;

    const int r     = blockIdx.x * 16 + l16;          // this lane's A/M row
    const int nbase = wave * 64;                      // 4 N-tiles per wave

    const float* Arow;
    if (a_is_btd) {
        int t = r >> 7, b = r & 127;                  // r = t*128 + b
        Arow = A + ((size_t)b * Tsz + t) * DIN;
    } else {
        Arow = A + (size_t)r * Psz;
    }
    const float* W0 = W + (size_t)(nbase + l16) * K;

    v8f acc[4] = {{}, {}, {}, {}};
    for (int k0 = 0; k0 < K; k0 += 4) {
        v2f av = *(const v2f*)(Arow + k0 + 2 * hi);   // A 16x4 f32 layout
#pragma unroll
        for (int j = 0; j < 4; ++j) {
            v2f bv = *(const v2f*)(W0 + (size_t)j * 16 * K + k0 + 2 * hi);
            acc[j] = __builtin_amdgcn_wmma_f32_16x16x4_f32(false, av, false, bv,
                                                           (short)0, acc[j], false, false);
        }
    }
    const int m0 = blockIdx.x * 16 + 8 * hi;
#pragma unroll
    for (int j = 0; j < 4; ++j) {
        const int n  = nbase + j * 16 + l16;
        const float bn = bias[n];
#pragma unroll
        for (int i = 0; i < 8; ++i)
            __builtin_nontemporal_store(acc[j][i] + bn, gx + (size_t)(m0 + i) * G4P + n);
    }
}

// ---------------------------------------------------------------------------
// LSTM scan: 8 workgroups, each owns 16 batch rows for all T steps.
// Wave w owns hidden slice p in [16w,16w+16): computes i,f,g,o tiles for that
// slice -> cell update is lane-local. W_hh kept in VGPRs as bf16 B-fragments.
// h exchanged through double-buffered LDS (bf16), one barrier per step.
// gx reads are software-pipelined one step ahead (registers) and prefetched
// 8 steps ahead into cache; gx reads are non-temporal.
// gx layout [T,B,512] f32; hout layout [T,B,128] f32 (temporal: lives in L2).
// ---------------------------------------------------------------------------
__global__ void __launch_bounds__(256, 1) lstm_scan_kernel(const float* __restrict__ gx,
                                                           const __bf16* __restrict__ Wb,
                                                           float* __restrict__ hout) {
    __shared__ __bf16 hbuf[2][16][136];               // padded rows vs bank conflicts

    const int lane = threadIdx.x & 31;
    const int wave = threadIdx.x >> 5;
    const int l16  = lane & 15;
    const int hi   = lane >> 4;
    const int b0   = blockIdx.x * 16;                 // batch slice
    const int p0   = wave * 16;                       // hidden slice of this wave

    // Preload W_hh B-fragments: gate g, k-chunk kk (K=128 -> 4 chunks of 32).
    // bf16 B 32x16: lane holds 16 consecutive K at column n, K base = kk*32 + hi*16.
    v16bf Bf[4][4];
#pragma unroll
    for (int g = 0; g < 4; ++g) {
        const int ncol = g * Psz + p0 + l16;
#pragma unroll
        for (int kk = 0; kk < 4; ++kk)
            Bf[g][kk] = *(const v16bf*)(Wb + (size_t)ncol * Psz + kk * 32 + hi * 16);
    }

    // Zero h_prev buffer 0
    for (int idx = threadIdx.x; idx < 16 * 136; idx += 256)
        (&hbuf[0][0][0])[idx] = (__bf16)0.0f;

    // gx gather for one timestep into registers (non-temporal: streamed once)
    auto load_gx = [&](v8f* dst, int t) {
        const size_t rowbase = ((size_t)t * Bsz + b0) * G4P;
#pragma unroll
        for (int g = 0; g < 4; ++g) {
            const int ncol = g * Psz + p0 + l16;
#pragma unroll
            for (int i = 0; i < 8; ++i)
                dst[g][i] = __builtin_nontemporal_load(
                    gx + rowbase + (size_t)(i + 8 * hi) * G4P + ncol);
        }
    };

    v8f gxr[4];
    load_gx(gxr, 0);                                  // prime the pipeline

    v8f c = {};                                       // cell state, lane-local
    __syncthreads();

    for (int t = 0; t < Tsz; ++t) {
        const int cur = t & 1, nxt = cur ^ 1;

        // A fragments from h_prev: bf16 A 16x32, per lane two 16B halves:
        // elems 0..15 = K kk*32 + hi*16 .. +15 (contiguous).
        v16bf Af[4];
#pragma unroll
        for (int kk = 0; kk < 4; ++kk) {
            const __bf16* base = &hbuf[cur][l16][kk * 32 + hi * 16];
            v8bf lo = *(const v8bf*)base;
            v8bf up = *(const v8bf*)(base + 8);
            Af[kk] = __builtin_shufflevector(lo, up, 0, 1, 2, 3, 4, 5, 6, 7,
                                             8, 9, 10, 11, 12, 13, 14, 15);
        }

        // Seed accumulators from the pipelined gx registers.
        v8f acc[4];
#pragma unroll
        for (int g = 0; g < 4; ++g) acc[g] = gxr[g];

        // Issue next step's gx loads now; they complete during WMMA+activations.
        if (t + 1 < Tsz) load_gx(gxr, t + 1);
        // Deep prefetch into cache: 256 threads x 128B cover the 32KB tile.
        if (t + 8 < Tsz) {
            const float* pf = gx + ((size_t)(t + 8) * Bsz + b0) * G4P
                            + (size_t)threadIdx.x * 32;
            __builtin_prefetch(pf, 0, 1);
        }

        // g += h_prev @ W_hh^T  (bf16 WMMA, f32 accumulate)
#pragma unroll
        for (int g = 0; g < 4; ++g)
#pragma unroll
            for (int kk = 0; kk < 4; ++kk)
                acc[g] = __builtin_amdgcn_wmma_f32_16x16x32_bf16(
                    false, Af[kk], false, Bf[g][kk], (short)0, acc[g], false, false);

        // Cell update (lane owns rows i+8*hi, column p0+l16)
#pragma unroll
        for (int i = 0; i < 8; ++i) {
            float iv = sigmoidf_(acc[0][i]);
            float fv = sigmoidf_(acc[1][i]);
            float gv = tanhf_(acc[2][i]);
            float ov = sigmoidf_(acc[3][i]);
            float cv = fv * c[i] + iv * gv;
            c[i] = cv;
            float hv = ov * tanhf_(cv);
            hbuf[nxt][i + 8 * hi][p0 + l16] = (__bf16)hv;
            hout[((size_t)t * Bsz + b0 + i + 8 * hi) * Psz + p0 + l16] = hv;
        }
        __syncthreads();   // nxt fully written; cur fully consumed
    }
}

// ---------------------------------------------------------------------------
// Final FC: out[b][t][o] = sum_p h1[t][b][p] * fc_w[o][p] + fc_b[o]
// f32 WMMA; block = 8 waves, each wave one M-tile x all 4 N-tiles (shared A).
// grid = (T*B)/32. h1 reads are temporal (resident in 192MB L2).
// ---------------------------------------------------------------------------
__global__ void __launch_bounds__(256) fc_kernel(const float* __restrict__ h1,
                                                 const float* __restrict__ fc_w,
                                                 const float* __restrict__ fc_b,
                                                 float* __restrict__ out) {
    const int lane = threadIdx.x & 31;
    const int wave = threadIdx.x >> 5;
    const int l16  = lane & 15;
    const int hi   = lane >> 4;

    const int mtile = blockIdx.x * 8 + wave;
    const int r     = mtile * 16 + l16;

    const float* Arow = h1 + (size_t)r * Psz;
    const float* W0   = fc_w + (size_t)l16 * Psz;     // N-tile j adds 16*Psz

    v8f acc[4] = {{}, {}, {}, {}};
    for (int k0 = 0; k0 < Psz; k0 += 4) {
        v2f av = *(const v2f*)(Arow + k0 + 2 * hi);
#pragma unroll
        for (int j = 0; j < 4; ++j) {
            v2f bv = *(const v2f*)(W0 + (size_t)j * 16 * Psz + k0 + 2 * hi);
            acc[j] = __builtin_amdgcn_wmma_f32_16x16x4_f32(false, av, false, bv,
                                                           (short)0, acc[j], false, false);
        }
    }
#pragma unroll
    for (int j = 0; j < 4; ++j) {
        const int n  = j * 16 + l16;
        const float bn = fc_b[n];
#pragma unroll
        for (int i = 0; i < 8; ++i) {
            int row = mtile * 16 + i + 8 * hi;        // row = t*128 + b
            int t = row >> 7, b = row & 127;
            out[((size_t)b * Tsz + t) * DOUT + n] = acc[j][i] + bn;
        }
    }
}

// ---------------------------------------------------------------------------
// Host launcher
// ---------------------------------------------------------------------------
extern "C" void kernel_launch(void* const* d_in, const int* in_sizes, int n_in,
                              void* d_out, int out_size, void* d_ws, size_t ws_size,
                              hipStream_t stream) {
    const float* x     = (const float*)d_in[0];
    const float* w_ih0 = (const float*)d_in[1];
    const float* w_hh0 = (const float*)d_in[2];
    const float* w_ih1 = (const float*)d_in[3];
    const float* w_hh1 = (const float*)d_in[4];
    const float* b_ih  = (const float*)d_in[5];
    const float* b_hh  = (const float*)d_in[6];
    const float* fc_w  = (const float*)d_in[7];
    const float* fc_b  = (const float*)d_in[8];
    float* out = (float*)d_out;

    char* ws = (char*)d_ws;
    const size_t GX_BYTES = (size_t)Tsz * Bsz * G4P * sizeof(float);   // 256 MiB
    const size_t H_BYTES  = (size_t)Tsz * Bsz * Psz * sizeof(float);   //  64 MiB
    float*  gx    = (float*)(ws);
    float*  h0    = (float*)(ws + GX_BYTES);
    float*  h1    = (float*)(ws + GX_BYTES + H_BYTES);
    __bf16* wb0   = (__bf16*)(ws + GX_BYTES + 2 * H_BYTES);
    __bf16* wb1   = wb0 + G4P * Psz;
    float*  bias0 = (float*)(wb1 + G4P * Psz);
    float*  bias1 = bias0 + G4P;

    // 1) weight/bias prep
    prep_kernel<<<256, 256, 0, stream>>>(w_hh0, w_hh1, b_ih, b_hh, wb0, wb1, bias0, bias1);

    // 2) layer-0 input GEMM: gx = x @ w_ih0^T + bias0   ([T,B,512])
    gemm_gx_kernel<<<(Tsz * Bsz) / 16, 256, 0, stream>>>(x, w_ih0, bias0, gx, DIN, 1);

    // 3) layer-0 recurrent scan -> h0 [T,B,128]
    lstm_scan_kernel<<<Bsz / 16, 256, 0, stream>>>(gx, wb0, h0);

    // 4) layer-1 input GEMM (reuses gx buffer): gx = h0 @ w_ih1^T + bias1
    gemm_gx_kernel<<<(Tsz * Bsz) / 16, 256, 0, stream>>>(h0, w_ih1, bias1, gx, Psz, 0);

    // 5) layer-1 recurrent scan -> h1 [T,B,128]
    lstm_scan_kernel<<<Bsz / 16, 256, 0, stream>>>(gx, wb1, h1);

    // 6) final FC -> out [B,T,64]
    fc_kernel<<<(Tsz * Bsz) / 32, 256, 0, stream>>>(h1, fc_w, fc_b, out);
}
